// SAST_block_6322191860267
// MI455X (gfx1250) — matrix operand order repaired and checked by reading, not compiled, not run on hardware
//
#include <hip/hip_runtime.h>
#include <hip/hip_bf16.h>

typedef __attribute__((ext_vector_type(16))) _Float16 v16h;
typedef __attribute__((ext_vector_type(8)))  float    v8f;
typedef __attribute__((ext_vector_type(4))) unsigned u32x4;
typedef __attribute__((ext_vector_type(8))) int      i32x8;
typedef __attribute__((ext_vector_type(4))) int      i32x4;

#if defined(__has_builtin)
#if __has_builtin(__builtin_amdgcn_tensor_load_to_lds)
#define HAVE_TDM 1
#endif
#endif
#ifndef HAVE_TDM
#define HAVE_TDM 0
#endif

// ---------------------------------------------------------------- WMMA helpers
__device__ __forceinline__ v8f wmma_f16(v16h a, v16h b, v8f c) {
  return __builtin_amdgcn_wmma_f32_16x16x32_f16(false, a, false, b, (short)0, c, false, false);
}

// A fragment: 16x32 f16, row-major source, leading dim ld (multiple of 8).
__device__ __forceinline__ v16h load_A(const _Float16* base, int ld) {
  int lane = threadIdx.x & 31;
  int row = lane & 15;
  int k0 = (lane < 16) ? 0 : 16;
  const _Float16* p = base + row * ld + k0;
  v16h a;
#pragma unroll
  for (int i = 0; i < 8; ++i) a[i] = p[i];
#pragma unroll
  for (int i = 0; i < 8; ++i) a[8 + i] = p[8 + i];
  return a;
}

// B fragment from TRANSPOSED (N-major) staging: base[n*ld + k] -> 2x ds_load_b128.
__device__ __forceinline__ v16h load_BT(const _Float16* base, int ld) {
  int lane = threadIdx.x & 31;
  int col = lane & 15;
  int k0 = (lane < 16) ? 0 : 16;
  const _Float16* p = base + col * ld + k0;
  v16h b;
#pragma unroll
  for (int i = 0; i < 16; ++i) b[i] = p[i];
  return b;
}

__device__ __forceinline__ unsigned pack2(float a, float b) {
  union { _Float16 h[2]; unsigned u; } x;
  x.h[0] = (_Float16)a; x.h[1] = (_Float16)b;
  return x.u;
}

#if HAVE_TDM
__device__ __forceinline__ unsigned lds_off(const void* p) {
  return (unsigned)(unsigned long long)p;  // LDS aperture: low 32 bits = LDS byte address
}
// 2D tile load via Tensor Data Mover. Units: DWORDs (data_size = 4B).
// OOB rows (>= tensor_rows) are zero-filled in LDS by the TDM.
__device__ __forceinline__ void tdm_load_2d(const void* gptr, unsigned ldsaddr,
                                            unsigned tile_dw, unsigned rows,
                                            unsigned tensor_rows, unsigned row_stride_dw,
                                            unsigned pad_en, unsigned pad_interval,
                                            unsigned pad_amount) {
  unsigned long long ga = (unsigned long long)gptr;
  u32x4 g0;
  g0[0] = 1u;                                                   // count=1 (valid descriptor)
  g0[1] = ldsaddr;                                              // lds_addr (bytes)
  g0[2] = (unsigned)(ga & 0xffffffffu);                         // global_addr[31:0]
  g0[3] = (unsigned)((ga >> 32) & 0x01ffffffu) | (2u << 30);    // global_addr[56:32] | type=2
  i32x8 g1;
  g1[0] = (int)((2u << 16) | (pad_en << 20) | (pad_interval << 22) | (pad_amount << 25));
  g1[1] = (int)(row_stride_dw << 16);                           // tensor_dim0 lo16
  g1[2] = (int)(((row_stride_dw >> 16) & 0xffffu) | ((tensor_rows & 0xffffu) << 16));
  g1[3] = (int)(((tensor_rows >> 16) & 0xffffu) | (tile_dw << 16));
  g1[4] = (int)(rows & 0xffffu);                                // tile_dim1 (tile_dim2=0)
  g1[5] = (int)row_stride_dw;                                   // tensor_dim0_stride lo32
  g1[6] = 0;
  g1[7] = 0;
  i32x4 z = {0, 0, 0, 0};
#if __clang_major__ >= 23
  i32x8 z8 = {0, 0, 0, 0, 0, 0, 0, 0};
  __builtin_amdgcn_tensor_load_to_lds(g0, g1, z, z, z8, 0);
#else
  __builtin_amdgcn_tensor_load_to_lds(g0, g1, z, z, 0);
#endif
}
#endif

// ---------------------------------------------------------------- K0: weight convert+transpose (once, ~1.5MB total)
// src: K x N fp32 row-major  ->  dst: N x K f16 row-major (N-major for B-tiles)
__global__ void k_cvtT(const float* __restrict__ src, _Float16* __restrict__ dst, int K, int N) {
  int idx = blockIdx.x * 256 + threadIdx.x;
  if (idx >= K * N) return;
  int k = idx / N, n = idx - k * N;  // coalesced read along n
  dst[(size_t)n * K + k] = (_Float16)src[idx];
}

// ---------------------------------------------------------------- K1: norm1
__global__ void k_norm1(const float* __restrict__ x, const float* __restrict__ g,
                        const float* __restrict__ b, float* __restrict__ X, int nrows) {
  int row = blockIdx.x * 8 + (threadIdx.x >> 5);
  if (row >= nrows) return;
  int lane = threadIdx.x & 31;
  const float4* xr = (const float4*)(x + (size_t)row * 256);
  union { float4 f4[2]; float v[8]; } u, gg, bb, oo;
  u.f4[0] = xr[lane * 2]; u.f4[1] = xr[lane * 2 + 1];
  float s = 0.f, s2 = 0.f;
#pragma unroll
  for (int i = 0; i < 8; ++i) { s += u.v[i]; s2 += u.v[i] * u.v[i]; }
#pragma unroll
  for (int off = 16; off; off >>= 1) { s += __shfl_xor(s, off, 32); s2 += __shfl_xor(s2, off, 32); }
  float mean = s * (1.f / 256.f);
  float rs = rsqrtf(s2 * (1.f / 256.f) - mean * mean + 1e-5f);
  gg.f4[0] = ((const float4*)g)[lane * 2]; gg.f4[1] = ((const float4*)g)[lane * 2 + 1];
  bb.f4[0] = ((const float4*)b)[lane * 2]; bb.f4[1] = ((const float4*)b)[lane * 2 + 1];
#pragma unroll
  for (int i = 0; i < 8; ++i) oo.v[i] = (u.v[i] - mean) * rs * gg.v[i] + bb.v[i];
  float4* Xr = (float4*)(X + (size_t)row * 256);
  Xr[lane * 2] = oo.f4[0]; Xr[lane * 2 + 1] = oo.f4[1];
}

// ---------------------------------------------------------------- K2: valid flags
__global__ void k_zero_flags(int* f, int n) {
  int i = blockIdx.x * 256 + threadIdx.x;
  if (i < n) f[i] = 0;
}
__global__ void k_set_flags(const int* __restrict__ asy, int* f, int n) {
  int i = blockIdx.x * 256 + threadIdx.x;
  if (i < n) f[asy[i]] = 1;
}

// ---------------------------------------------------------------- K3: LN2 + tokenize
__global__ void k_ln2_tok(const float* __restrict__ X, const int* __restrict__ itok,
                          const int* __restrict__ iwin, const int* __restrict__ flag,
                          const float* __restrict__ g, const float* __restrict__ b,
                          _Float16* __restrict__ tok, float* __restrict__ acc,
                          int R, int Km) {
  int t = blockIdx.x * 8 + (threadIdx.x >> 5);
  if (t >= R) return;
  int lane = threadIdx.x & 31;
  int m = t / Km;
  int p = itok[t];
  _Float16* tr = tok + (size_t)t * 256;
  float* ar = acc + (size_t)t * 256;
  if (!flag[p]) {
    uint4 z4 = {0u, 0u, 0u, 0u};
    ((uint4*)tr)[lane] = z4;
    float4 zf = {0.f, 0.f, 0.f, 0.f};
    ((float4*)ar)[lane * 2] = zf; ((float4*)ar)[lane * 2 + 1] = zf;
    return;
  }
  int grow = iwin[m] * 64 + (p & 63);
  const float4* xr = (const float4*)(X + (size_t)grow * 256);
  union { float4 f4[2]; float v[8]; } u, gg, bb, oo;
  u.f4[0] = xr[lane * 2]; u.f4[1] = xr[lane * 2 + 1];
  float s = 0.f, s2 = 0.f;
#pragma unroll
  for (int i = 0; i < 8; ++i) { s += u.v[i]; s2 += u.v[i] * u.v[i]; }
#pragma unroll
  for (int off = 16; off; off >>= 1) { s += __shfl_xor(s, off, 32); s2 += __shfl_xor(s2, off, 32); }
  float mean = s * (1.f / 256.f);
  float rs = rsqrtf(s2 * (1.f / 256.f) - mean * mean + 1e-5f);
  gg.f4[0] = ((const float4*)g)[lane * 2]; gg.f4[1] = ((const float4*)g)[lane * 2 + 1];
  bb.f4[0] = ((const float4*)b)[lane * 2]; bb.f4[1] = ((const float4*)b)[lane * 2 + 1];
  union { _Float16 h[8]; uint4 u4; } th;
#pragma unroll
  for (int i = 0; i < 8; ++i) { oo.v[i] = (u.v[i] - mean) * rs * gg.v[i] + bb.v[i]; th.h[i] = (_Float16)oo.v[i]; }
  ((uint4*)tr)[lane] = th.u4;
  ((float4*)ar)[lane * 2] = oo.f4[0]; ((float4*)ar)[lane * 2 + 1] = oo.f4[1];
}

// ---------------------------------------------------------------- K4: QKV GEMM (A + double-buffered B via TDM)
__global__ void k_qkv(const _Float16* __restrict__ tok, const _Float16* __restrict__ WT,
                      const float* __restrict__ bias, _Float16* __restrict__ out, int R) {
  __shared__ _Float16 As[64 * 264];    // 512B rows + 16B TDM padding
  __shared__ _Float16 Bs[2][64 * 48];  // N-major, 64B rows + 32B TDM padding
  int tid = threadIdx.x, wave = tid >> 5, lane = tid & 31;
  int row0 = blockIdx.x * 64, col0 = blockIdx.y * 64;
  int wr = wave & 3, wcg = wave >> 2;
  bool full = (row0 + 64 <= R);
  v8f accv[2] = {};
#if HAVE_TDM
  if (wave == 0) {
    tdm_load_2d(tok + (size_t)row0 * 256, lds_off(As), 128, 64, (unsigned)(R - row0), 128, 1, 6, 3);
    tdm_load_2d(WT + (size_t)col0 * 256, lds_off(&Bs[0][0]), 16, 64, 64, 128, 1, 3, 7);
  }
  for (int kk = 0; kk < 8; ++kk) {
    if (wave == 0) {
      if (kk < 7)
        tdm_load_2d(WT + (size_t)col0 * 256 + (kk + 1) * 32, lds_off(&Bs[(kk + 1) & 1][0]),
                    16, 64, 64, 128, 1, 3, 7);
      if (kk < 7) __builtin_amdgcn_s_wait_tensorcnt(1);
      else        __builtin_amdgcn_s_wait_tensorcnt(0);
    }
    __syncthreads();
    v16h af = load_A(&As[wr * 16 * 264 + kk * 32], 264);
#pragma unroll
    for (int bc = 0; bc < 2; ++bc) {
      v16h bf = load_BT(&Bs[kk & 1][(wcg * 2 + bc) * 16 * 48], 48);
      accv[bc] = wmma_f16(af, bf, accv[bc]);
    }
    __syncthreads();
  }
#else
  {
    unsigned* As32 = (unsigned*)As;
    const unsigned* tok32 = (const unsigned*)tok;
#pragma unroll
    for (int ii = 0; ii < 32; ++ii) {
      int i = tid + ii * 256;
      int r = i >> 7, cp = i & 127;
      int gr = row0 + r;
      As32[r * 132 + cp] = (full || gr < R) ? tok32[(size_t)gr * 128 + cp] : 0u;
    }
  }
  for (int kk = 0; kk < 8; ++kk) {
    unsigned* Bs32 = (unsigned*)(&Bs[0][0]);
    const unsigned* WT32 = (const unsigned*)WT;
#pragma unroll
    for (int ii = 0; ii < 4; ++ii) {
      int i = tid + ii * 256;
      int c = i >> 4, rp = i & 15;
      Bs32[c * 24 + rp] = WT32[(size_t)(col0 + c) * 128 + kk * 16 + rp];
    }
    __syncthreads();
    v16h af = load_A(&As[wr * 16 * 264 + kk * 32], 264);
#pragma unroll
    for (int bc = 0; bc < 2; ++bc) {
      v16h bf = load_BT(&Bs[0][(wcg * 2 + bc) * 16 * 48], 48);
      accv[bc] = wmma_f16(af, bf, accv[bc]);
    }
    __syncthreads();
  }
#endif
  int coln = lane & 15, rb = (lane < 16) ? 0 : 8;
  if (full) {
#pragma unroll
    for (int bc = 0; bc < 2; ++bc) {
      int ct = col0 + (wcg * 2 + bc) * 16 + coln;
      float bv = bias[ct];
      size_t base = (size_t)(row0 + wr * 16 + rb) * 768 + ct;
#pragma unroll
      for (int r = 0; r < 8; ++r) out[base + (size_t)r * 768] = (_Float16)(accv[bc][r] + bv);
    }
  } else {
#pragma unroll
    for (int bc = 0; bc < 2; ++bc) {
      int ct = col0 + (wcg * 2 + bc) * 16 + coln;
      float bv = bias[ct];
#pragma unroll
      for (int r = 0; r < 8; ++r) {
        int gr = row0 + wr * 16 + rb + r;
        if (gr < R) out[(size_t)gr * 768 + ct] = (_Float16)(accv[bc][r] + bv);
      }
    }
  }
}

// ---------------------------------------------------------------- K5: per-(window, head-pair) fused attention
__global__ void k_attn(const _Float16* __restrict__ qkv, const int* __restrict__ itok,
                       const int* __restrict__ flag, _Float16* __restrict__ aout) {
  __shared__ _Float16 Qs[2][48 * 40];
  __shared__ _Float16 Ks[2][48 * 40];
  __shared__ _Float16 Vs[2][64 * 32];
  __shared__ float    Ss[2][48 * 48];
  __shared__ _Float16 Ps[2][48 * 72];
  __shared__ int fl[48];
  int m = blockIdx.x;
  int wave = threadIdx.x >> 5, lane = threadIdx.x & 31;
  int h = blockIdx.y * 2 + wave;
  if (threadIdx.x < 48) fl[threadIdx.x] = flag[itok[m * 48 + threadIdx.x]];
  __syncthreads();
  unsigned* Vs32 = (unsigned*)(&Vs[wave][0]);
#if HAVE_TDM
  {
    const char* gbase = (const char*)qkv + ((size_t)m * 48 * 768 + (size_t)h * 96) * 2;
    tdm_load_2d(gbase,       lds_off(&Qs[wave][0]), 16, 48, 48, 384, 1, 3, 3);
    tdm_load_2d(gbase + 64,  lds_off(&Ks[wave][0]), 16, 48, 48, 384, 1, 3, 3);
    tdm_load_2d(gbase + 128, lds_off(&Vs[wave][0]), 16, 48, 48, 384, 0, 0, 0);
    __builtin_amdgcn_s_wait_tensorcnt(0);
  }
#else
  {
    const unsigned* b32 = (const unsigned*)qkv + ((size_t)m * 48 * 768 + h * 96) / 2;
    unsigned* Qs32 = (unsigned*)(&Qs[wave][0]);
    unsigned* Ks32 = (unsigned*)(&Ks[wave][0]);
    for (int r = 0; r < 48; ++r) {
      unsigned w0 = b32[(size_t)r * 384 + lane];
      if (lane < 16) Qs32[r * 20 + lane] = w0;
      else           Ks32[r * 20 + (lane - 16)] = w0;
      if (lane < 16) Vs32[r * 16 + lane] = b32[(size_t)r * 384 + 32 + lane];
    }
  }
#endif
  for (int r = 48; r < 64; ++r)
    if (lane < 16) Vs32[r * 16 + lane] = 0u;

  const float scale = 0.17677669529663687f;  // 32^-0.5
  int coln = lane & 15, k0b = (lane < 16) ? 0 : 16, rb = (lane < 16) ? 0 : 8;
#pragma unroll
  for (int i = 0; i < 3; ++i) {
    v16h aq = load_A(&Qs[wave][i * 16 * 40], 40);
#pragma unroll
    for (int j = 0; j < 3; ++j) {
      v16h bk;
      const _Float16* kp = &Ks[wave][(j * 16 + coln) * 40 + k0b];
#pragma unroll
      for (int e = 0; e < 16; ++e) bk[e] = kp[e];
      v8f s = {};
      s = wmma_f16(aq, bk, s);
#pragma unroll
      for (int r = 0; r < 8; ++r) Ss[wave][(i * 16 + rb + r) * 48 + j * 16 + coln] = s[r] * scale;
    }
  }
  for (int r = lane; r < 48; r += 32) {
    float mx = -1e30f;
    for (int c = 0; c < 48; ++c) if (fl[c]) mx = fmaxf(mx, Ss[wave][r * 48 + c]);
    float sum = 0.f;
    for (int c = 0; c < 48; ++c) if (fl[c]) sum += __expf(Ss[wave][r * 48 + c] - mx);
    float inv = 1.f / sum;
    for (int c = 0; c < 48; ++c)
      Ps[wave][r * 72 + c] = fl[c] ? (_Float16)(__expf(Ss[wave][r * 48 + c] - mx) * inv) : (_Float16)0.f;
    for (int c = 48; c < 64; ++c) Ps[wave][r * 72 + c] = (_Float16)0.f;
  }
#pragma unroll
  for (int i = 0; i < 3; ++i) {
    v16h ap0 = load_A(&Ps[wave][i * 16 * 72], 72);
    v16h ap1 = load_A(&Ps[wave][i * 16 * 72 + 32], 72);
#pragma unroll
    for (int j = 0; j < 2; ++j) {
      v16h bv0, bv1;
#pragma unroll
      for (int e = 0; e < 16; ++e) bv0[e] = Vs[wave][(k0b + e) * 32 + j * 16 + coln];
#pragma unroll
      for (int e = 0; e < 16; ++e) bv1[e] = Vs[wave][(32 + k0b + e) * 32 + j * 16 + coln];
      v8f o = {};
      o = wmma_f16(ap0, bv0, o);
      o = wmma_f16(ap1, bv1, o);
#pragma unroll
      for (int r = 0; r < 8; ++r)
        aout[(size_t)(m * 48 + i * 16 + rb + r) * 256 + h * 32 + j * 16 + coln] = (_Float16)o[r];
    }
  }
}

// ---------------------------------------------------------------- K6: proj GEMM fused residual (TDM A + B)
__global__ void k_proj(const _Float16* __restrict__ attn, const _Float16* __restrict__ WT,
                       const float* __restrict__ pb, const float* __restrict__ ls1,
                       float* __restrict__ acc, int R) {
  __shared__ _Float16 As[64 * 264];
  __shared__ _Float16 Bs[2][64 * 48];
  int tid = threadIdx.x, wave = tid >> 5, lane = tid & 31;
  int row0 = blockIdx.x * 64, col0 = blockIdx.y * 64;
  int wr = wave & 3, wcg = wave >> 2;
  bool full = (row0 + 64 <= R);
  v8f accv[2] = {};
#if HAVE_TDM
  if (wave == 0) {
    tdm_load_2d(attn + (size_t)row0 * 256, lds_off(As), 128, 64, (unsigned)(R - row0), 128, 1, 6, 3);
    tdm_load_2d(WT + (size_t)col0 * 256, lds_off(&Bs[0][0]), 16, 64, 64, 128, 1, 3, 7);
  }
  for (int kk = 0; kk < 8; ++kk) {
    if (wave == 0) {
      if (kk < 7)
        tdm_load_2d(WT + (size_t)col0 * 256 + (kk + 1) * 32, lds_off(&Bs[(kk + 1) & 1][0]),
                    16, 64, 64, 128, 1, 3, 7);
      if (kk < 7) __builtin_amdgcn_s_wait_tensorcnt(1);
      else        __builtin_amdgcn_s_wait_tensorcnt(0);
    }
    __syncthreads();
    v16h af = load_A(&As[wr * 16 * 264 + kk * 32], 264);
#pragma unroll
    for (int bc = 0; bc < 2; ++bc) {
      v16h bf = load_BT(&Bs[kk & 1][(wcg * 2 + bc) * 16 * 48], 48);
      accv[bc] = wmma_f16(af, bf, accv[bc]);
    }
    __syncthreads();
  }
#else
  {
    unsigned* As32 = (unsigned*)As;
    const unsigned* a32 = (const unsigned*)attn;
#pragma unroll
    for (int ii = 0; ii < 32; ++ii) {
      int i = tid + ii * 256;
      int r = i >> 7, cp = i & 127;
      int gr = row0 + r;
      As32[r * 132 + cp] = (full || gr < R) ? a32[(size_t)gr * 128 + cp] : 0u;
    }
  }
  for (int kk = 0; kk < 8; ++kk) {
    unsigned* Bs32 = (unsigned*)(&Bs[0][0]);
    const unsigned* WT32 = (const unsigned*)WT;
#pragma unroll
    for (int ii = 0; ii < 4; ++ii) {
      int i = tid + ii * 256;
      int c = i >> 4, rp = i & 15;
      Bs32[c * 24 + rp] = WT32[(size_t)(col0 + c) * 128 + kk * 16 + rp];
    }
    __syncthreads();
    v16h af = load_A(&As[wr * 16 * 264 + kk * 32], 264);
#pragma unroll
    for (int bc = 0; bc < 2; ++bc) {
      v16h bf = load_BT(&Bs[0][(wcg * 2 + bc) * 16 * 48], 48);
      accv[bc] = wmma_f16(af, bf, accv[bc]);
    }
    __syncthreads();
  }
#endif
  int coln = lane & 15, rb = (lane < 16) ? 0 : 8;
  if (full) {
#pragma unroll
    for (int bc = 0; bc < 2; ++bc) {
      int ct = col0 + (wcg * 2 + bc) * 16 + coln;
      float bv = pb[ct], l1 = ls1[ct];
      size_t base = (size_t)(row0 + wr * 16 + rb) * 256 + ct;
#pragma unroll
      for (int r = 0; r < 8; ++r) {
        size_t o = base + (size_t)r * 256;
        acc[o] = acc[o] + l1 * (accv[bc][r] + bv);
      }
    }
  } else {
#pragma unroll
    for (int bc = 0; bc < 2; ++bc) {
      int ct = col0 + (wcg * 2 + bc) * 16 + coln;
      float bv = pb[ct], l1 = ls1[ct];
#pragma unroll
      for (int r = 0; r < 8; ++r) {
        int gr = row0 + wr * 16 + rb + r;
        if (gr < R) {
          size_t o = (size_t)gr * 256 + ct;
          acc[o] = acc[o] + l1 * (accv[bc][r] + bv);
        }
      }
    }
  }
}

// ---------------------------------------------------------------- K7: fused MLP (W1/W2 chunks via TDM)
__global__ void k_mlp(const float* __restrict__ acc, const _Float16* __restrict__ W1T,
                      const float* __restrict__ b1, const _Float16* __restrict__ W2T,
                      const float* __restrict__ b2, float* __restrict__ y2, int R) {
  __shared__ _Float16 Ys[64 * 264];
  __shared__ _Float16 Wb[256 * 48];  // W1 chunks: 32 x ld272 (8704); W2 chunks: 256 x ld48 (12288)
  __shared__ _Float16 Hb[64 * 40];
  int tid = threadIdx.x;
  int wave = tid >> 5, lane = tid & 31;
  int row0 = blockIdx.x * 64;
  bool full = (row0 + 64 <= R);
  unsigned* Ys32 = (unsigned*)Ys;
  const float2* a2 = (const float2*)acc;
#pragma unroll
  for (int ii = 0; ii < 32; ++ii) {
    int i = tid + ii * 256;
    int r = i >> 7, cp = i & 127;
    int gr = row0 + r;
    if (full || gr < R) {
      float2 v = a2[(size_t)gr * 128 + cp];
      Ys32[r * 132 + cp] = pack2(v.x, v.y);
    } else Ys32[r * 132 + cp] = 0u;
  }
  v8f oacc[4][2] = {};
  int hr = wave & 3, hcid = wave >> 2;
  int coln = lane & 15, rb = (lane < 16) ? 0 : 8;
  for (int hc = 0; hc < 32; ++hc) {
    __syncthreads();
#if HAVE_TDM
    if (wave == 0) {
      // W1 chunk: rows n = hc*32..+31 of W1T (1024 x 256), ld=272 in LDS
      tdm_load_2d(W1T + (size_t)hc * 32 * 256, lds_off(Wb), 128, 32, 32, 128, 1, 6, 7);
      __builtin_amdgcn_s_wait_tensorcnt(0);
    }
#else
    {
      unsigned* Wb32 = (unsigned*)Wb;
      const unsigned* W1T32 = (const unsigned*)W1T;
#pragma unroll
      for (int ii = 0; ii < 16; ++ii) {
        int i = tid + ii * 256;
        int n = i >> 7, kp = i & 127;
        Wb32[n * 136 + kp] = W1T32[(size_t)(hc * 32 + n) * 128 + kp];
      }
    }
#endif
    __syncthreads();
    {
      v8f hacc = {};
#pragma unroll
      for (int kk = 0; kk < 8; ++kk) {
        v16h af = load_A(&Ys[hr * 16 * 264 + kk * 32], 264);
        v16h bf = load_BT(&Wb[hcid * 16 * 272 + kk * 32], 272);
        hacc = wmma_f16(af, bf, hacc);
      }
      float bb = b1[hc * 32 + hcid * 16 + coln];
#pragma unroll
      for (int r = 0; r < 8; ++r) {
        float xg = hacc[r] + bb;
        float th = tanhf(0.7978845608f * (xg + 0.044715f * xg * xg * xg));
        Hb[(hr * 16 + rb + r) * 40 + hcid * 16 + coln] = (_Float16)(0.5f * xg * (1.f + th));
      }
    }
    __syncthreads();
#if HAVE_TDM
    if (wave == 0) {
      // W2 chunk: all 256 rows of W2T (256 x 1024), cols k = hc*32..+31, ld=48 in LDS
      tdm_load_2d(W2T + (size_t)hc * 32, lds_off(Wb), 16, 256, 256, 512, 1, 3, 7);
      __builtin_amdgcn_s_wait_tensorcnt(0);
    }
#else
    {
      unsigned* Wb32 = (unsigned*)Wb;
      const unsigned* W2T32 = (const unsigned*)W2T;
#pragma unroll
      for (int ii = 0; ii < 16; ++ii) {
        int i = tid + ii * 256;
        int n = i >> 3, kp = i & 7;
        Wb32[n * 24 + kp] = W2T32[(size_t)n * 512 + hc * 16 + kp];
      }
    }
#endif
    __syncthreads();
#pragma unroll
    for (int rt = 0; rt < 4; ++rt) {
      v16h af = load_A(&Hb[rt * 16 * 40], 40);
#pragma unroll
      for (int cc = 0; cc < 2; ++cc) {
        v16h bf = load_BT(&Wb[(wave + cc * 8) * 16 * 48], 48);
        oacc[rt][cc] = wmma_f16(af, bf, oacc[rt][cc]);
      }
    }
  }
  if (full) {
#pragma unroll
    for (int rt = 0; rt < 4; ++rt)
#pragma unroll
      for (int cc = 0; cc < 2; ++cc) {
        int ct = (wave + cc * 8) * 16 + coln;
        float bb = b2[ct];
        size_t base = (size_t)(row0 + rt * 16 + rb) * 256 + ct;
#pragma unroll
        for (int r = 0; r < 8; ++r) y2[base + (size_t)r * 256] = oacc[rt][cc][r] + bb;
      }
  } else {
#pragma unroll
    for (int rt = 0; rt < 4; ++rt)
#pragma unroll
      for (int cc = 0; cc < 2; ++cc) {
        int ct = (wave + cc * 8) * 16 + coln;
        float bb = b2[ct];
#pragma unroll
        for (int r = 0; r < 8; ++r) {
          int gr = row0 + rt * 16 + rb + r;
          if (gr < R) y2[(size_t)gr * 256 + ct] = oacc[rt][cc][r] + bb;
        }
      }
  }
}

// ---------------------------------------------------------------- K8: deterministic reductions
__global__ void k_wsum(const float* __restrict__ y2, const int* __restrict__ itok,
                       const int* __restrict__ flag, float* __restrict__ wsum, int Km) {
  int m = blockIdx.x, c = threadIdx.x;
  float s = 0.f;
  for (int k = 0; k < Km; ++k) {
    int t = m * Km + k;
    if (flag[itok[t]]) s += y2[(size_t)t * 256 + c];
  }
  wsum[(size_t)m * 256 + c] = s;
}
__global__ void k_bsum(const float* __restrict__ wsum, const int* __restrict__ iwin,
                       const int* __restrict__ Bp, float* __restrict__ bsums, int Mn, int NW) {
  int b = blockIdx.x, c = threadIdx.x;
  int Bv = *Bp;
  if (b >= Bv) return;
  int nwb = NW / Bv;
  float s = 0.f;
  for (int m = 0; m < Mn; ++m)
    if (iwin[m] / nwb == b) s += wsum[(size_t)m * 256 + c];
  bsums[b * 256 + c] = s;
}

// ---------------------------------------------------------------- K9: blend + ls2 residual + scatter
__global__ void k_final(const float* __restrict__ y2, const float* __restrict__ acc,
                        const float* __restrict__ bsums, const float* __restrict__ ls2,
                        const int* __restrict__ itok, const int* __restrict__ iwin,
                        const int* __restrict__ flag, const int* __restrict__ Bp,
                        const int* __restrict__ CBp, float* __restrict__ X,
                        int R, int Km, int NW) {
  int t = blockIdx.x * 8 + (threadIdx.x >> 5);
  if (t >= R) return;
  int p = itok[t];
  if (!flag[p]) return;
  int lane = threadIdx.x & 31;
  int m = t / Km;
  int Bv = *Bp, CB = *CBp;
  int nwb = NW / Bv;
  int b = iwin[m] / nwb;
  float invD = 0.5f / (float)(nwb * 64);
  const float4* y4 = (const float4*)(y2 + (size_t)t * 256);
  const float4* a4 = (const float4*)(acc + (size_t)t * 256);
  const float4* s4 = (const float4*)(bsums + (size_t)b * 256);
  const float4* l4 = (const float4*)ls2;
  float4* X4 = (float4*)(X + ((size_t)iwin[m] * 64 + (p & 63)) * 256);
#pragma unroll
  for (int j = 0; j < 2; ++j) {
    int idx = lane * 2 + j;
    union { float4 f4; float v[4]; } yv, av, sv, lv, ov;
    yv.f4 = y4[idx]; av.f4 = a4[idx]; lv.f4 = l4[idx];
    if (CB) {
      sv.f4 = s4[idx];
#pragma unroll
      for (int q = 0; q < 4; ++q) yv.v[q] = 0.5f * yv.v[q] + invD * sv.v[q];
    }
#pragma unroll
    for (int q = 0; q < 4; ++q) ov.v[q] = av.v[q] + lv.v[q] * yv.v[q];
    X4[idx] = ov.f4;
  }
}

// ---------------------------------------------------------------- host
extern "C" void kernel_launch(void* const* d_in, const int* in_sizes, int n_in,
                              void* d_out, int out_size, void* d_ws, size_t ws_size,
                              hipStream_t stream) {
  const float* x    = (const float*)d_in[0];
  const int* iwin   = (const int*)d_in[1];
  const int* itok   = (const int*)d_in[2];
  const int* asy    = (const int*)d_in[4];
  const int* Bp     = (const int*)d_in[6];
  const int* CBp    = (const int*)d_in[7];
  const float* n1g  = (const float*)d_in[8];
  const float* n1b  = (const float*)d_in[9];
  const float* n2g  = (const float*)d_in[10];
  const float* n2b  = (const float*)d_in[11];
  const float* qkvw = (const float*)d_in[12];
  const float* qkvb = (const float*)d_in[13];
  const float* pw   = (const float*)d_in[14];
  const float* pbv  = (const float*)d_in[15];
  const float* ls1  = (const float*)d_in[16];
  const float* w1   = (const float*)d_in[17];
  const float* b1   = (const float*)d_in[18];
  const float* w2   = (const float*)d_in[19];
  const float* b2   = (const float*)d_in[20];
  const float* ls2  = (const float*)d_in[21];
  float* X = (float*)d_out;

  const int C = in_sizes[8];
  const int Mn = in_sizes[1];
  const int Km = in_sizes[2] / Mn;
  const int n_asy = in_sizes[4];
  const long nrows = (long)in_sizes[0] / C;
  const int NW = (int)(nrows / 64);
  const int R = Mn * Km;

  char* p = (char*)d_ws;
  auto alloc = [&](size_t bytes) { void* r = (void*)p; p += (bytes + 255) & ~(size_t)255; return r; };
  int* flag       = (int*)alloc((size_t)Mn * 64 * 4);
  _Float16* tok   = (_Float16*)alloc((size_t)R * C * 2);
  float* accb     = (float*)alloc((size_t)R * C * 4);
  _Float16* qkvB  = (_Float16*)alloc((size_t)R * 3 * C * 2);
  float* wsum     = (float*)alloc((size_t)Mn * C * 4);
  float* bsums    = (float*)alloc((size_t)64 * C * 4);
  _Float16* qkvWT = (_Float16*)alloc((size_t)3 * C * C * 2);   // 768 x 256 f16
  _Float16* projWT= (_Float16*)alloc((size_t)C * C * 2);       // 256 x 256 f16
  _Float16* W1T   = (_Float16*)alloc((size_t)4 * C * C * 2);   // 1024 x 256 f16
  _Float16* W2T   = (_Float16*)alloc((size_t)4 * C * C * 2);   // 256 x 1024 f16
  float* y2 = (float*)qkvB;
  (void)ws_size; (void)n_in; (void)out_size;

  // one-time weight convert+transpose (L2-resident afterwards)
  k_cvtT<<<dim3((C * 3 * C + 255) / 256), 256, 0, stream>>>(qkvw, qkvWT, C, 3 * C);
  k_cvtT<<<dim3((C * C + 255) / 256), 256, 0, stream>>>(pw, projWT, C, C);
  k_cvtT<<<dim3((C * 4 * C + 255) / 256), 256, 0, stream>>>(w1, W1T, C, 4 * C);
  k_cvtT<<<dim3((4 * C * C + 255) / 256), 256, 0, stream>>>(w2, W2T, 4 * C, C);

  k_norm1<<<dim3((unsigned)((nrows + 7) / 8)), 256, 0, stream>>>(x, n1g, n1b, X, (int)nrows);
  k_zero_flags<<<dim3((Mn * 64 + 255) / 256), 256, 0, stream>>>(flag, Mn * 64);
  k_set_flags<<<dim3((n_asy + 255) / 256), 256, 0, stream>>>(asy, flag, n_asy);
  k_ln2_tok<<<dim3((R + 7) / 8), 256, 0, stream>>>(X, itok, iwin, flag, n2g, n2b, tok, accb, R, Km);
  k_qkv<<<dim3((R + 63) / 64, 12), 256, 0, stream>>>(tok, qkvWT, qkvb, qkvB, R);
  k_attn<<<dim3(Mn, 4), 64, 0, stream>>>(qkvB, itok, flag, tok);
  k_proj<<<dim3((R + 63) / 64, 4), 256, 0, stream>>>(tok, projWT, pbv, ls1, accb, R);
  k_mlp<<<dim3((R + 63) / 64), 256, 0, stream>>>(accb, W1T, b1, W2T, b2, y2, R);
  k_wsum<<<dim3(Mn), 256, 0, stream>>>(y2, itok, flag, wsum, Km);
  k_bsum<<<dim3(64), 256, 0, stream>>>(wsum, iwin, Bp, bsums, Mn, NW);
  k_final<<<dim3((R + 7) / 8), 256, 0, stream>>>(y2, accb, bsums, ls2, itok, iwin, flag, Bp, CBp, X, R, Km, NW);
}
